// SVGABlock_3289944949108
// MI455X (gfx1250) — matrix operand order repaired and checked by reading, not compiled, not run on hardware
//
#include <hip/hip_runtime.h>
#include <hip/hip_bf16.h>
#include <cmath>

// ---------------------------------------------------------------------------
// SVGA block for MI455X (gfx1250): bf16 WMMA, fused GEMM chains through LDS,
// parity-min MRConv, async global->LDS tile staging.
// ---------------------------------------------------------------------------

typedef float  v8f   __attribute__((ext_vector_type(8)));
typedef __bf16 v16bf __attribute__((ext_vector_type(16)));
typedef int    v4i_vs __attribute__((vector_size(16)));   // matches builtin proto

union Frag32B { v16bf v; uint4 u[2]; };

#define EPI_BN        0
#define EPI_BN_GELU   1
#define EPI_BN_RES    2
#define EPI_GELU_BIAS 3
#define EPI_BIAS      4

static constexpr int MT     = 64;          // pixels per workgroup M-tile
static constexpr int HWIMG  = 112 * 112;   // 12544 (divisible by MT)
static constexpr int BASE_C = 96;
static constexpr int MTOT   = 16 * HWIMG;  // 200704
static constexpr int NBLK   = MTOT / MT;   // 3136

// ---- async global->LDS staging (CDNA5), with safe fallback ----------------
#if defined(__HIP_DEVICE_COMPILE__) && defined(__has_builtin)
#  if __has_builtin(__builtin_amdgcn_global_load_async_to_lds_b128)
#    define USE_ASYNC_STAGE 1
#  endif
#endif
#ifndef USE_ASYNC_STAGE
#  define USE_ASYNC_STAGE 0
#endif

__device__ __forceinline__ void wait_asynccnt0() {
#if defined(__HIP_DEVICE_COMPILE__)
#  if defined(__has_builtin) && __has_builtin(__builtin_amdgcn_s_wait_asynccnt)
    __builtin_amdgcn_s_wait_asynccnt(0);
#  else
    asm volatile("s_wait_asynccnt 0" ::: "memory");
#  endif
#endif
}

#if USE_ASYNC_STAGE
__device__ __forceinline__ void async_copy16(const char* g, char* l) {
    __builtin_amdgcn_global_load_async_to_lds_b128(
        (__attribute__((address_space(1))) v4i_vs*)(uintptr_t)g,
        (__attribute__((address_space(3))) v4i_vs*)l, 0, 0);
}
#endif

__device__ __forceinline__ float gelu_exact(float x) {
    return 0.5f * x * (1.0f + erff(x * 0.70710678118654752f));
}

// ---------------------------------------------------------------------------
// Weight pre-swizzle: pack W[N][K] (fp32) into per-lane WMMA B-fragment order.
// b[i] of lane L in tile (kt,nt):  K = kt*32 + (i&7) + 16*(i>=8) + 8*(L>=16),  N = nt*16 + (L&15)
// ---------------------------------------------------------------------------
__global__ void prep_wfrag(const float* __restrict__ w, __bf16* __restrict__ dst,
                           int K, int N) {
    int total = K * N;
    int NTl = N / 16;
    for (int e = blockIdx.x * blockDim.x + threadIdx.x; e < total;
         e += gridDim.x * blockDim.x) {
        int i    = e & 15;
        int lane = (e >> 4) & 31;
        int rest = e >> 9;
        int nt = rest % NTl;
        int kt = rest / NTl;
        int k = kt * 32 + (i & 7) + ((i >> 3) << 4) + ((lane >= 16) ? 8 : 0);
        int n = nt * 16 + (lane & 15);
        dst[e] = (__bf16)w[(size_t)n * K + k];
    }
}

// ---------------------------------------------------------------------------
// Stage a [64][CIN] bf16 channels-last tile from global into LDS (stride LDA).
// ---------------------------------------------------------------------------
template<int CIN, int LDA>
__device__ __forceinline__ void stage_tile_bf16(const __bf16* __restrict__ src,
                                                char* Asm, int pbase, int tid) {
    constexpr int ROWB   = CIN * 2;
    constexpr int CHUNKS = ROWB / 16;
    const char* gsrc = (const char*)src + (size_t)pbase * ROWB;
    for (int e = tid; e < MT * CHUNKS; e += 256) {
        int p = e / CHUNKS;
        int q = e - p * CHUNKS;
        const char* g = gsrc + (size_t)p * ROWB + q * 16;
        char*       l = Asm + (size_t)p * (LDA * 2) + q * 16;
#if USE_ASYNC_STAGE
        async_copy16(g, l);
#else
        *(uint4*)l = *(const uint4*)g;
#endif
    }
#if USE_ASYNC_STAGE
    wait_asynccnt0();
#endif
}

// ---------------------------------------------------------------------------
// One GEMM phase: A from LDS [64][LDA] bf16, B fragments from global (pre-
// swizzled), epilogue into LDS (bf16 [64][OSTRIDE] or f32 [NOUT][72]).
// 8 waves: 4 M-subtiles x 2 N interleave groups. v_wmma_f32_16x16x32_bf16.
// ---------------------------------------------------------------------------
template<int CIN, int LDA, int NOUT, int EPI, bool OF32, int OSTRIDE>
__device__ __forceinline__ void gemm_phase(
    const char* __restrict__ Asm, char* __restrict__ Osm,
    const __bf16* __restrict__ wfrag,
    const float* __restrict__ gam, const float* __restrict__ bet,
    const float* __restrict__ mu,  const float* __restrict__ var,
    const float* __restrict__ bias,
    const float* __restrict__ res,
    int b, int pimg, int tid)
{
    const int lane  = tid & 31;
    const int wv    = tid >> 5;
    const int msub  = wv & 3;
    const int nhalf = wv >> 2;
    const int khalf = (lane < 16) ? 0 : 8;       // A-frag K offset per lane half
    const int prow  = msub * 16 + (lane & 15);   // A-frag pixel row
    constexpr int NT = NOUT / 16;
    constexpr int KT = CIN / 32;
    const char* arow0 = Asm + ((size_t)prow * LDA + khalf) * 2;

    for (int nt = nhalf; nt < NT; nt += 2) {
        v8f acc = {0.f, 0.f, 0.f, 0.f, 0.f, 0.f, 0.f, 0.f};
        #pragma unroll
        for (int kt = 0; kt < KT; ++kt) {
            Frag32B a, bw;
            const char* arow = arow0 + kt * 64;    // 32 bf16 per K-tile
            a.u[0] = *(const uint4*)(arow);        // K = kt*32+khalf    .. +7
            a.u[1] = *(const uint4*)(arow + 32);   // K = kt*32+khalf+16 .. +23
            const __bf16* bp = wfrag + (size_t)(kt * NT + nt) * 512 + lane * 16;
            bw.u[0] = *(const uint4*)(bp);
            bw.u[1] = *(const uint4*)(bp + 8);
            acc = __builtin_amdgcn_wmma_f32_16x16x32_bf16(false, a.v, false, bw.v,
                                                          (short)0, acc, false, false);
        }
        // ---- epilogue: per-lane column n, 8 pixel rows ----
        const int n = nt * 16 + (lane & 15);
        float scale, shift;
        if constexpr (EPI == EPI_BN || EPI == EPI_BN_GELU || EPI == EPI_BN_RES) {
            float s = gam[n] * rsqrtf(var[n] + 1e-5f);
            scale = s;
            shift = (bias[n] - mu[n]) * s + bet[n];
        } else {
            scale = 1.f;
            shift = bias[n];
        }
        const int poff = msub * 16 + ((lane < 16) ? 0 : 8);
        #pragma unroll
        for (int r = 0; r < 8; ++r) {
            float val = acc[r] * scale + shift;
            if constexpr (EPI == EPI_BN_GELU || EPI == EPI_GELU_BIAS)
                val = gelu_exact(val);
            int pp = poff + r;
            if constexpr (EPI == EPI_BN_RES)
                val += res[((size_t)b * BASE_C + n) * (size_t)HWIMG + pimg + pp];
            if constexpr (OF32) {
                ((float*)Osm)[(size_t)n * 72 + pp] = val;
            } else {
                ((__bf16*)Osm)[(size_t)pp * OSTRIDE + n] = (__bf16)val;
            }
        }
    }
}

// ---- cooperative coalesced stores -----------------------------------------
template<int NOUT>
__device__ __forceinline__ void store_cl(const char* __restrict__ Osm,
                                         char* __restrict__ gout,
                                         int pbase, int cstride, int choff, int tid) {
    constexpr int CHO = NOUT * 2 / 16;
    for (int e = tid; e < MT * CHO; e += 256) {
        int p = e / CHO;
        int q = e - p * CHO;
        *(uint4*)(gout + ((size_t)(pbase + p) * cstride + choff) * 2 + q * 16) =
            *(const uint4*)(Osm + (size_t)p * (NOUT * 2) + q * 16);
    }
}

__device__ __forceinline__ void store_nchw96(const char* __restrict__ Osm,
                                             float* __restrict__ gout,
                                             int b, int pimg, int tid) {
    const float (*Os)[72] = (const float (*)[72])Osm;
    for (int e = tid; e < 96 * 16; e += 256) {     // 16 float4 per 64-px row
        int n = e >> 4;
        int q = e & 15;
        *(float4*)(gout + ((size_t)b * 96 + n) * (size_t)HWIMG + pimg + q * 4) =
            *(const float4*)&Os[n][q * 4];
    }
}

// ---------------------------------------------------------------------------
// fc1: x (fp32 NCHW) -> hcat[:, 0:96] bf16 channels-last, BN folded
// ---------------------------------------------------------------------------
__global__ __launch_bounds__(256) void fc1_kernel(
    const float* __restrict__ x, const __bf16* __restrict__ wfrag,
    const float* __restrict__ g1, const float* __restrict__ be1,
    const float* __restrict__ m1, const float* __restrict__ v1,
    const float* __restrict__ b1, __bf16* __restrict__ hcat)
{
    extern __shared__ char smem_[];
    char* Asm = smem_;                      // [64][104] bf16 = 13312
    char* Osm = smem_ + MT * 104 * 2;       // [64][96]  bf16 = 12288
    const int tid = threadIdx.x;
    const int pbase = blockIdx.x * MT;
    const int b = pbase / HWIMG;
    const int pimg = pbase - b * HWIMG;

    __bf16 (*As)[104] = (__bf16 (*)[104])Asm;
    for (int e = tid; e < MT * 96; e += 256) {
        int c = e >> 6;                     // MT == 64
        int p = e & 63;
        As[p][c] = (__bf16)x[((size_t)b * 96 + c) * (size_t)HWIMG + pimg + p];
    }
    __syncthreads();
    gemm_phase<96, 104, 96, EPI_BN, false, 96>(
        Asm, Osm, wfrag, g1, be1, m1, v1, b1, nullptr, b, pimg, tid);
    __syncthreads();
    store_cl<96>(Osm, (char*)hcat, pbase, 192, 0, tid);
}

// ---------------------------------------------------------------------------
// Fused grapher: hcat(192) --BN+GELU--> [LDS] --BN+res--> y(96)
// ---------------------------------------------------------------------------
__global__ __launch_bounds__(256) void fused_grapher(
    const __bf16* __restrict__ hcat,
    const __bf16* __restrict__ wfragG, const __bf16* __restrict__ wfrag2,
    const float* __restrict__ gg, const float* __restrict__ beg,
    const float* __restrict__ mg, const float* __restrict__ vg,
    const float* __restrict__ bg,
    const float* __restrict__ g2, const float* __restrict__ be2,
    const float* __restrict__ m2, const float* __restrict__ v2,
    const float* __restrict__ b2,
    const float* __restrict__ res,
    __bf16* __restrict__ ybuf)
{
    extern __shared__ char smem_[];
    char* Asm = smem_;                      // [64][200] bf16 = 25600
    char* Gsm = smem_ + 25600;              // [64][200] bf16 = 25600
    char* Osm = smem_ + 51200;              // [64][96]  bf16 = 12288
    const int tid = threadIdx.x;
    const int pbase = blockIdx.x * MT;
    const int b = pbase / HWIMG;
    const int pimg = pbase - b * HWIMG;

    stage_tile_bf16<192, 200>(hcat, Asm, pbase, tid);
    __syncthreads();
    gemm_phase<192, 200, 192, EPI_BN_GELU, false, 200>(
        Asm, Gsm, wfragG, gg, beg, mg, vg, bg, nullptr, b, pimg, tid);
    __syncthreads();
    gemm_phase<192, 200, 96, EPI_BN_RES, false, 96>(
        Gsm, Osm, wfrag2, g2, be2, m2, v2, b2, res, b, pimg, tid);
    __syncthreads();
    store_cl<96>(Osm, (char*)ybuf, pbase, 96, 0, tid);
}

// ---------------------------------------------------------------------------
// Fused FFN: y(96) --bias+GELU--> [LDS 384] --bias--> out (fp32 NCHW)
// ---------------------------------------------------------------------------
__global__ __launch_bounds__(256) void fused_ffn(
    const __bf16* __restrict__ ybuf,
    const __bf16* __restrict__ wfragF1, const __bf16* __restrict__ wfragF2,
    const float* __restrict__ bf1, const float* __restrict__ bf2,
    float* __restrict__ out)
{
    extern __shared__ char smem_[];
    char* Asm = smem_;                      // [64][104] bf16 = 13312
    char* Fsm = smem_ + 13312;              // [64][392] bf16 = 50176
    char* Osm = smem_ + 63488;              // [96][72]  f32  = 27648
    const int tid = threadIdx.x;
    const int pbase = blockIdx.x * MT;
    const int b = pbase / HWIMG;
    const int pimg = pbase - b * HWIMG;

    stage_tile_bf16<96, 104>(ybuf, Asm, pbase, tid);
    __syncthreads();
    gemm_phase<96, 104, 384, EPI_GELU_BIAS, false, 392>(
        Asm, Fsm, wfragF1, nullptr, nullptr, nullptr, nullptr, bf1, nullptr,
        b, pimg, tid);
    __syncthreads();
    gemm_phase<384, 392, 96, EPI_BIAS, true, 72>(
        Fsm, Osm, wfragF2, nullptr, nullptr, nullptr, nullptr, bf2, nullptr,
        b, pimg, tid);
    __syncthreads();
    store_nchw96(Osm, out, b, pimg, tid);
}

// ---------------------------------------------------------------------------
// MRConv (k=2): per-parity (min1, argmin, min2) along W (rows) and H (cols),
// then xj = max(0, x - colMinExcl, x - rowMinExcl); h in hcat ch 0..95.
// ---------------------------------------------------------------------------
__global__ __launch_bounds__(96) void row_stats_kernel(const __bf16* __restrict__ hcat,
                                                       float4* __restrict__ rstats) {
    int bh = blockIdx.x;                 // b*112 + h
    int c  = threadIdx.x;                // 0..95
    int b  = bh / 112;
    int h  = bh - b * 112;
    const __bf16* base = hcat + ((size_t)(b * HWIMG + h * 112)) * 192 + c;
    float m1e = 3.4e38f, m2e = 3.4e38f, m1o = 3.4e38f, m2o = 3.4e38f;
    int ae = -1, ao = -1;
    for (int w = 0; w < 112; ++w) {
        float v = (float)base[(size_t)w * 192];
        if (w & 1) {
            if (v < m1o) { m2o = m1o; m1o = v; ao = w; } else if (v < m2o) m2o = v;
        } else {
            if (v < m1e) { m2e = m1e; m1e = v; ae = w; } else if (v < m2e) m2e = v;
        }
    }
    rstats[((size_t)bh * 2 + 0) * 96 + c] = make_float4(m1e, m2e, __int_as_float(ae), 0.f);
    rstats[((size_t)bh * 2 + 1) * 96 + c] = make_float4(m1o, m2o, __int_as_float(ao), 0.f);
}

__global__ __launch_bounds__(96) void col_stats_kernel(const __bf16* __restrict__ hcat,
                                                       float4* __restrict__ cstats) {
    int bw = blockIdx.x;                 // b*112 + w
    int c  = threadIdx.x;
    int b  = bw / 112;
    int w  = bw - b * 112;
    const __bf16* base = hcat + ((size_t)b * HWIMG + w) * 192 + c;
    float m1e = 3.4e38f, m2e = 3.4e38f, m1o = 3.4e38f, m2o = 3.4e38f;
    int ae = -1, ao = -1;
    for (int h = 0; h < 112; ++h) {
        float v = (float)base[(size_t)h * 112 * 192];
        if (h & 1) {
            if (v < m1o) { m2o = m1o; m1o = v; ao = h; } else if (v < m2o) m2o = v;
        } else {
            if (v < m1e) { m2e = m1e; m1e = v; ae = h; } else if (v < m2e) m2e = v;
        }
    }
    cstats[((size_t)bw * 2 + 0) * 96 + c] = make_float4(m1e, m2e, __int_as_float(ae), 0.f);
    cstats[((size_t)bw * 2 + 1) * 96 + c] = make_float4(m1o, m2o, __int_as_float(ao), 0.f);
}

__global__ __launch_bounds__(96) void apply_xj_kernel(__bf16* __restrict__ hcat,
                                                      const float4* __restrict__ rstats,
                                                      const float4* __restrict__ cstats) {
    int bh = blockIdx.x;
    int c  = threadIdx.x;
    int b  = bh / 112;
    int h  = bh - b * 112;
    float4 re = rstats[((size_t)bh * 2 + 0) * 96 + c];
    float4 ro = rstats[((size_t)bh * 2 + 1) * 96 + c];
    __bf16* base = hcat + ((size_t)(b * HWIMG + h * 112)) * 192;
    int hp = h & 1;
    for (int w = 0; w < 112; ++w) {
        float v = (float)base[(size_t)w * 192 + c];
        float4 rs = (w & 1) ? ro : re;
        float rowExcl = (w == __float_as_int(rs.z)) ? rs.y : rs.x;
        float4 cs = cstats[(((size_t)(b * 112 + w)) * 2 + hp) * 96 + c];
        float colExcl = (h == __float_as_int(cs.z)) ? cs.y : cs.x;
        float xj = fmaxf(0.f, fmaxf(v - colExcl, v - rowExcl));
        base[(size_t)w * 192 + 96 + c] = (__bf16)xj;    // hcat channels 96..191
    }
}

// ---------------------------------------------------------------------------
// Workspace layout (bytes)
// ---------------------------------------------------------------------------
static constexpr size_t OFF_HC = 0;                                  // [M][192] bf16
static constexpr size_t OFF_Y  = (size_t)MTOT * 192 * 2;             // [M][96] bf16
static constexpr size_t OFF_RS = OFF_Y + (size_t)MTOT * 96 * 2;      // row stats
static constexpr size_t OFF_CS = OFF_RS + (size_t)1792 * 2 * 96 * 16;
static constexpr size_t OFF_WF = OFF_CS + (size_t)1792 * 2 * 96 * 16;

// wfrag sub-offsets (bf16 elements)
static constexpr size_t WF_W1  = 0;                 // 96x96
static constexpr size_t WF_WG  = WF_W1 + 96 * 96;   // 192x192
static constexpr size_t WF_W2  = WF_WG + 192 * 192; // 192->96
static constexpr size_t WF_WF1 = WF_W2 + 192 * 96;  // 96->384
static constexpr size_t WF_WF2 = WF_WF1 + 96 * 384; // 384->96

static constexpr int SM_FC1 = MT * 104 * 2 + MT * 96 * 2;            // 25600
static constexpr int SM_GR  = 25600 + 25600 + MT * 96 * 2;           // 63488
static constexpr int SM_FF  = 13312 + 50176 + 96 * 72 * 4;           // 91136

extern "C" void kernel_launch(void* const* d_in, const int* in_sizes, int n_in,
                              void* d_out, int out_size, void* d_ws, size_t ws_size,
                              hipStream_t stream) {
    (void)in_sizes; (void)n_in; (void)out_size; (void)ws_size;

    const float* x   = (const float*)d_in[0];
    const float* w1  = (const float*)d_in[1];
    const float* b1  = (const float*)d_in[2];
    const float* g1  = (const float*)d_in[3];
    const float* be1 = (const float*)d_in[4];
    const float* m1  = (const float*)d_in[5];
    const float* v1  = (const float*)d_in[6];
    const float* wg  = (const float*)d_in[7];
    const float* bg  = (const float*)d_in[8];
    const float* gg  = (const float*)d_in[9];
    const float* beg = (const float*)d_in[10];
    const float* mg  = (const float*)d_in[11];
    const float* vg  = (const float*)d_in[12];
    const float* w2  = (const float*)d_in[13];
    const float* b2  = (const float*)d_in[14];
    const float* g2  = (const float*)d_in[15];
    const float* be2 = (const float*)d_in[16];
    const float* m2  = (const float*)d_in[17];
    const float* v2  = (const float*)d_in[18];
    const float* wf1 = (const float*)d_in[19];
    const float* bf1 = (const float*)d_in[20];
    const float* wf2 = (const float*)d_in[21];
    const float* bf2 = (const float*)d_in[22];

    char* ws = (char*)d_ws;
    __bf16* hcat  = (__bf16*)(ws + OFF_HC);
    __bf16* ybuf  = (__bf16*)(ws + OFF_Y);
    float4* rstat = (float4*)(ws + OFF_RS);
    float4* cstat = (float4*)(ws + OFF_CS);
    __bf16* wf    = (__bf16*)(ws + OFF_WF);

    // allow >64KB dynamic LDS where needed (WGP has 320KB)
    (void)hipFuncSetAttribute(reinterpret_cast<const void*>(&fused_ffn),
                              hipFuncAttributeMaxDynamicSharedMemorySize, 131072);
    (void)hipFuncSetAttribute(reinterpret_cast<const void*>(&fused_grapher),
                              hipFuncAttributeMaxDynamicSharedMemorySize, 131072);

    // ---- weight fragment pre-swizzle (fp32 -> bf16, WMMA B-layout) ----
    prep_wfrag<<<(96 * 96 + 255) / 256, 256, 0, stream>>>(w1, wf + WF_W1, 96, 96);
    prep_wfrag<<<(192 * 192 + 255) / 256, 256, 0, stream>>>(wg, wf + WF_WG, 192, 192);
    prep_wfrag<<<(192 * 96 + 255) / 256, 256, 0, stream>>>(w2, wf + WF_W2, 192, 96);
    prep_wfrag<<<(96 * 384 + 255) / 256, 256, 0, stream>>>(wf1, wf + WF_WF1, 96, 384);
    prep_wfrag<<<(384 * 96 + 255) / 256, 256, 0, stream>>>(wf2, wf + WF_WF2, 384, 96);

    // ---- fc1: x (fp32 NCHW) -> hcat[:, 0:96], BN folded ----
    fc1_kernel<<<NBLK, 256, SM_FC1, stream>>>(x, wf + WF_W1, g1, be1, m1, v1, b1, hcat);

    // ---- MRConv xj -> hcat[:, 96:192] ----
    row_stats_kernel<<<16 * 112, 96, 0, stream>>>(hcat, rstat);
    col_stats_kernel<<<16 * 112, 96, 0, stream>>>(hcat, cstat);
    apply_xj_kernel<<<16 * 112, 96, 0, stream>>>(hcat, rstat, cstat);

    // ---- fused grapher: hcat(192) -> GELU(192) [LDS] -> +res -> y(96) ----
    fused_grapher<<<NBLK, 256, SM_GR, stream>>>(
        hcat, wf + WF_WG, wf + WF_W2,
        gg, beg, mg, vg, bg, g2, be2, m2, v2, b2, x, ybuf);

    // ---- fused FFN: y(96) -> GELU(384) [LDS] -> out (fp32 NCHW) ----
    fused_ffn<<<NBLK, 256, SM_FF, stream>>>(
        ybuf, wf + WF_WF1, wf + WF_WF2, bf1, bf2, (float*)d_out);
}
